// simple_GAT_regression_59502476919379
// MI455X (gfx1250) — matrix operand order repaired and checked by reading, not compiled, not run on hardware
//
#include <hip/hip_runtime.h>
#include <math.h>

#define NN   50000      // nodes
#define NE   800000     // raw edges
#define NET  850000     // edges + self loops
#define FIN  32
#define HID  96
#define NG   64
#define BN_EPS 1e-5f
#define LN_EPS 1e-5f
#define NEG_SLOPE 0.2f

typedef __attribute__((ext_vector_type(16))) __bf16    v16bf;
typedef __attribute__((ext_vector_type(8)))  float     v8f;
typedef __attribute__((ext_vector_type(4)))  unsigned  u32x4;

struct Q2 { u32x4 a, b; };   // 32B = one v16bf fragment

// ---------- helpers ----------
// round-to-nearest-even f32 -> bf16 bits
__device__ __forceinline__ unsigned short rne_bf16(float f) {
    unsigned u = __float_as_uint(f);
    return (unsigned short)((u + 0x7FFFu + ((u >> 16) & 1u)) >> 16);
}
// split f32 into hi+lo bf16 bit patterns (hi RNE, lo = RNE(f - hi)): ~16 mantissa bits total
__device__ __forceinline__ void split_bits(float f, unsigned short& hb, unsigned short& lb) {
    hb = rne_bf16(f);
    float fh = __uint_as_float(((unsigned)hb) << 16);
    lb = rne_bf16(f - fh);                    // f - fh exactly representable
}
// order-preserving float<->uint for atomicMax on signed floats
__device__ __forceinline__ unsigned f2ord(float f) {
    unsigned u = __float_as_uint(f);
    return (u & 0x80000000u) ? ~u : (u | 0x80000000u);
}
__device__ __forceinline__ float ord2f(unsigned u) {
    unsigned v = (u & 0x80000000u) ? (u & 0x7FFFFFFFu) : ~u;
    return __uint_as_float(v);
}
__device__ __forceinline__ void edge_sd(const int* __restrict__ ei, int e, int& s, int& d) {
    if (e < NE) { s = ei[e]; d = ei[NE + e]; }
    else        { s = d = e - NE; }           // self loop
}
// load one 32-byte WMMA fragment from two 16B-aligned chunks of packed bf16
__device__ __forceinline__ v16bf load_frag(const unsigned short* p0, const unsigned short* p1) {
    Q2 q;
    q.a = *(const u32x4*)p0;
    q.b = *(const u32x4*)p1;
    return __builtin_bit_cast(v16bf, q);
}

// ---------- 0a) split W [K,96] f32 -> transposed bf16 hi/lo Wt[96][K] ----------
__global__ void k_split_w(const float* __restrict__ W, unsigned short* __restrict__ whi,
                          unsigned short* __restrict__ wlo, int K) {
    int idx = blockIdx.x * blockDim.x + threadIdx.x;
    if (idx >= HID * K) return;
    int n = idx / K, k = idx % K;
    unsigned short hb, lb;
    split_bits(W[k * HID + n], hb, lb);
    whi[n * K + k] = hb;
    wlo[n * K + k] = lb;
}

// ---------- 0b) split layer-0 input X [N,32] f32 -> bf16 hi/lo ----------
__global__ void k_split_x(const float* __restrict__ X, unsigned short* __restrict__ xhi,
                          unsigned short* __restrict__ xlo) {
    int idx = blockIdx.x * blockDim.x + threadIdx.x;
    if (idx >= NN * FIN) return;
    unsigned short hb, lb;
    split_bits(X[idx], hb, lb);
    xhi[idx] = hb;
    xlo[idx] = lb;
}

// ---------- 1) GEMM: H[N,96] = X[N,K] @ W[K,96] via split-precision bf16 WMMA ----------
// Xhi/Xlo: row-major [N,K] bf16; Whi/Wlo: transposed [96,K] bf16.
template <int K>
__global__ void k_gemm(const unsigned short* __restrict__ Xhi,
                       const unsigned short* __restrict__ Xlo,
                       const unsigned short* __restrict__ Whi,
                       const unsigned short* __restrict__ Wlo,
                       float* __restrict__ H) {
    const int rt   = blockIdx.x * 4 + (threadIdx.x >> 5);   // 16-row tile, wave-uniform
    if (rt >= NN / 16) return;
    const int lane = threadIdx.x & 31;
    const int m    = lane & 15;
    const int half = lane >> 4;
    const int row  = rt * 16 + m;
    const int kbA  = half * 16;    // A: K-base for this lane half
    const int kbB  = half * 16;    // B: K-base for this lane half

    v8f acc[6];
#pragma unroll
    for (int ct = 0; ct < 6; ++ct) acc[ct] = (v8f){0.f,0.f,0.f,0.f,0.f,0.f,0.f,0.f};

#pragma unroll
    for (int k0 = 0; k0 < K; k0 += 32) {
        const unsigned short* xh = Xhi + (size_t)row * K + k0 + kbA;
        const unsigned short* xl = Xlo + (size_t)row * K + k0 + kbA;
        v16bf ahi = load_frag(xh, xh + 8);      // 16 consecutive K values
        v16bf alo = load_frag(xl, xl + 8);
#pragma unroll
        for (int ct = 0; ct < 6; ++ct) {
            const unsigned short* wh = Whi + (size_t)(ct * 16 + m) * K + k0 + kbB;
            const unsigned short* wl = Wlo + (size_t)(ct * 16 + m) * K + k0 + kbB;
            v16bf bhi = load_frag(wh, wh + 8);  // 16 consecutive K values
            v16bf blo = load_frag(wl, wl + 8);
            acc[ct] = __builtin_amdgcn_wmma_f32_16x16x32_bf16(false, ahi, false, bhi,
                                                              (short)0, acc[ct], false, false);
            acc[ct] = __builtin_amdgcn_wmma_f32_16x16x32_bf16(false, ahi, false, blo,
                                                              (short)0, acc[ct], false, false);
            acc[ct] = __builtin_amdgcn_wmma_f32_16x16x32_bf16(false, alo, false, bhi,
                                                              (short)0, acc[ct], false, false);
        }
    }
    // C/D layout: VGPR j -> M = half*8 + j, N = lane%16
#pragma unroll
    for (int ct = 0; ct < 6; ++ct)
#pragma unroll
        for (int j = 0; j < 8; ++j)
            H[(size_t)(rt * 16 + half * 8 + j) * HID + ct * 16 + m] = acc[ct][j];
}

// ---------- 2) per-node attention scalars: as = h . a_src, ad = h . a_dst ----------
__global__ void k_attn(const float* __restrict__ h, const float* __restrict__ asrc,
                       const float* __restrict__ adst, float* __restrict__ as_,
                       float* __restrict__ ad_) {
    const int node = (blockIdx.x * blockDim.x + threadIdx.x) >> 5;  // wave per node
    if (node >= NN) return;
    const int lane = threadIdx.x & 31;
    const float* row = h + (size_t)node * HID;
    float sv = 0.f, dv = 0.f;
    for (int c = lane; c < HID; c += 32) { float v = row[c]; sv += v * asrc[c]; dv += v * adst[c]; }
    for (int off = 16; off; off >>= 1) { sv += __shfl_down(sv, off); dv += __shfl_down(dv, off); }
    if (lane == 0) { as_[node] = sv; ad_[node] = dv; }
}

// ---------- 3) per-layer init ----------
__global__ void k_init_layer(unsigned* __restrict__ menc, float* __restrict__ s,
                             float* __restrict__ agg) {
    int idx = blockIdx.x * blockDim.x + threadIdx.x;
    if (idx < NN * HID) agg[idx] = 0.f;
    if (idx < NN) { menc[idx] = 0u; s[idx] = 0.f; }   // ord(0) < ord(any finite float)
}

// ---------- 4) edge logits + segment max ----------
__global__ void k_edge_max(const int* __restrict__ ei, const float* __restrict__ as_,
                           const float* __restrict__ ad_, float* __restrict__ elog,
                           unsigned* __restrict__ menc) {
    int e = blockIdx.x * blockDim.x + threadIdx.x;
    if (e >= NET) return;
    int s, d; edge_sd(ei, e, s, d);
    float l = as_[s] + ad_[d];
    l = (l >= 0.f) ? l : NEG_SLOPE * l;
    elog[e] = l;
    atomicMax(&menc[d], f2ord(l));
}

// ---------- 5) exp + segment sum ----------
__global__ void k_edge_exp(const int* __restrict__ ei, float* __restrict__ elog,
                           const unsigned* __restrict__ menc, float* __restrict__ s) {
    int e = blockIdx.x * blockDim.x + threadIdx.x;
    if (e >= NET) return;
    int si, d; edge_sd(ei, e, si, d); (void)si;
    float ex = expf(elog[e] - ord2f(menc[d]));
    elog[e] = ex;
    atomicAdd(&s[d], ex);
}

// ---------- 6) normalize weights ----------
__global__ void k_edge_norm(const int* __restrict__ ei, float* __restrict__ elog,
                            const float* __restrict__ s) {
    int e = blockIdx.x * blockDim.x + threadIdx.x;
    if (e >= NET) return;
    int si, d; edge_sd(ei, e, si, d); (void)si;
    elog[e] = elog[e] / (s[d] + 1e-16f);
}

// ---------- 7) weighted scatter-add: agg[dst,c] += h[src,c] * w[e] ----------
__global__ void k_edge_agg(const int* __restrict__ ei, const float* __restrict__ elog,
                           const float* __restrict__ h, float* __restrict__ agg) {
    int idx = blockIdx.x * blockDim.x + threadIdx.x;
    if (idx >= NET * HID) return;
    int e = idx / HID, c = idx % HID;
    int s, d; edge_sd(ei, e, s, d);
    atomicAdd(&agg[(size_t)d * HID + c], h[(size_t)s * HID + c] * elog[e]);
}

// ---------- 8) bias + relu + BN(eval); also emit split-bf16 copy for next GEMM ----------
__global__ void k_epilogue(const float* __restrict__ agg, const float* __restrict__ cb,
                           const float* __restrict__ g, const float* __restrict__ bb,
                           const float* __restrict__ rm, const float* __restrict__ rv,
                           float* __restrict__ xcur, unsigned short* __restrict__ xhi,
                           unsigned short* __restrict__ xlo) {
    int idx = blockIdx.x * blockDim.x + threadIdx.x;
    if (idx >= NN * HID) return;
    int c = idx % HID;
    float v = agg[idx] + cb[c];
    v = fmaxf(v, 0.f);
    v = (v - rm[c]) * rsqrtf(rv[c] + BN_EPS) * g[c] + bb[c];
    xcur[idx] = v;
    unsigned short hb, lb;
    split_bits(v, hb, lb);
    xhi[idx] = hb;
    xlo[idx] = lb;
}

// ---------- 9) pooling ----------
__global__ void k_zero_pool(float* __restrict__ gsum, float* __restrict__ gcnt) {
    int idx = blockIdx.x * blockDim.x + threadIdx.x;
    if (idx < NG * HID) gsum[idx] = 0.f;
    if (idx < NG) gcnt[idx] = 0.f;
}
__global__ void k_pool(const float* __restrict__ xcur, const int* __restrict__ batch,
                       float* __restrict__ gsum, float* __restrict__ gcnt) {
    int idx = blockIdx.x * blockDim.x + threadIdx.x;
    if (idx >= NN * HID) return;
    int i = idx / HID, c = idx % HID;
    int b = batch[i];
    atomicAdd(&gsum[b * HID + c], xcur[idx]);
    if (c == 0) atomicAdd(&gcnt[b], 1.f);
}

// ---------- 10) regression head: one block (96 threads) per graph ----------
__global__ void k_head(const float* __restrict__ gsum, const float* __restrict__ gcnt,
                       const float* __restrict__ ftW, const float* __restrict__ ftb,
                       const float* __restrict__ lng, const float* __restrict__ lnb,
                       const float* __restrict__ a1W, const float* __restrict__ a1b,
                       const float* __restrict__ a2W, const float* __restrict__ a2b,
                       const float* __restrict__ rW,  const float* __restrict__ rb,
                       float* __restrict__ out) {
    const int g = blockIdx.x, c = threadIdx.x;    // c in [0,96)
    __shared__ float feat[HID], t[HID], u[24], red[HID], sc[2];
    feat[c] = gsum[g * HID + c] / fmaxf(gcnt[g], 1.f);
    out[NG + g * HID + c] = feat[c];              // feat output (second return value)
    __syncthreads();
    float tc = ftb[c];
    for (int k = 0; k < HID; ++k) tc += feat[k] * ftW[k * HID + c];
    red[c] = tc;
    __syncthreads();
    if (c == 0) {
        float mu = 0.f; for (int k = 0; k < HID; ++k) mu += red[k];
        mu /= HID;
        float var = 0.f; for (int k = 0; k < HID; ++k) { float d = red[k] - mu; var += d * d; }
        var /= HID;
        sc[0] = mu; sc[1] = rsqrtf(var + LN_EPS);
    }
    __syncthreads();
    tc = (tc - sc[0]) * sc[1] * lng[c] + lnb[c];
    tc = fmaxf(tc, 0.f);
    t[c] = tc;
    __syncthreads();
    if (c < 24) {
        float uc = a1b[c];
        for (int k = 0; k < HID; ++k) uc += t[k] * a1W[k * 24 + c];
        u[c] = fmaxf(uc, 0.f);
    }
    __syncthreads();
    if (c == 0) {
        float z = a2b[0];
        for (int k = 0; k < 24; ++k) z += u[k] * a2W[k];
        sc[0] = 1.f / (1.f + expf(-z));           // attention weight
    }
    __syncthreads();
    red[c] = t[c] * sc[0] * rW[c];
    __syncthreads();
    if (c == 0) {
        float p = rb[0]; for (int k = 0; k < HID; ++k) p += red[k];
        out[g] = p;                               // pred output (first return value)
    }
}

// ---------- host ----------
static inline int cdiv(int a, int b) { return (a + b - 1) / b; }

extern "C" void kernel_launch(void* const* d_in, const int* in_sizes, int n_in,
                              void* d_out, int out_size, void* d_ws, size_t ws_size,
                              hipStream_t stream) {
    (void)n_in; (void)out_size; (void)ws_size;
    // --- resolve input indices (insertion-order vs sorted-key flattening) ---
    int iX = 0, iEdge, iBatch;
    if (in_sizes[1] == 2 * NE) { iEdge = 1; iBatch = 2; }       // X, edge_index, batch
    else                       { iBatch = 1; iEdge = 2; }       // X, batch, edge_index
    const float* X  = (const float*)d_in[iX];
    const int*   ei = (const int*)d_in[iEdge];
    const int*   batch = (const int*)d_in[iBatch];

    const float *cW[5], *cAs[5], *cAd[5], *cB[5];
    const float *bG[5], *bB[5], *bRm[5], *bRv[5];
    const float *ftW, *ftb, *lng, *lnb, *a1W, *a1b, *a2W, *a2b, *rW, *rb;
    const int base = 3;
    if (in_sizes[base] == FIN * HID) {
        // insertion order: convs (W, a_src, a_dst, b) x5, bns (g, b, rm, rv) x5, head
        for (int l = 0; l < 5; ++l) {
            cW[l]  = (const float*)d_in[base + 4 * l + 0];
            cAs[l] = (const float*)d_in[base + 4 * l + 1];
            cAd[l] = (const float*)d_in[base + 4 * l + 2];
            cB[l]  = (const float*)d_in[base + 4 * l + 3];
            bG[l]  = (const float*)d_in[base + 20 + 4 * l + 0];
            bB[l]  = (const float*)d_in[base + 20 + 4 * l + 1];
            bRm[l] = (const float*)d_in[base + 20 + 4 * l + 2];
            bRv[l] = (const float*)d_in[base + 20 + 4 * l + 3];
        }
        int h0 = base + 40;
        ftW = (const float*)d_in[h0 + 0]; ftb = (const float*)d_in[h0 + 1];
        lng = (const float*)d_in[h0 + 2]; lnb = (const float*)d_in[h0 + 3];
        a1W = (const float*)d_in[h0 + 4]; a1b = (const float*)d_in[h0 + 5];
        a2W = (const float*)d_in[h0 + 6]; a2b = (const float*)d_in[h0 + 7];
        rW  = (const float*)d_in[h0 + 8]; rb  = (const float*)d_in[h0 + 9];
    } else {
        // sorted keys: bns (b,g,rm,rv) x5, convs (W, a_dst, a_src, b) x5, head sorted
        for (int l = 0; l < 5; ++l) {
            bB[l]  = (const float*)d_in[base + 4 * l + 0];
            bG[l]  = (const float*)d_in[base + 4 * l + 1];
            bRm[l] = (const float*)d_in[base + 4 * l + 2];
            bRv[l] = (const float*)d_in[base + 4 * l + 3];
            cW[l]  = (const float*)d_in[base + 20 + 4 * l + 0];
            cAd[l] = (const float*)d_in[base + 20 + 4 * l + 1];
            cAs[l] = (const float*)d_in[base + 20 + 4 * l + 2];
            cB[l]  = (const float*)d_in[base + 20 + 4 * l + 3];
        }
        int h0 = base + 40;  // at1_W, at1_b, at2_W, at2_b, ft_W, ft_b, ln_b, ln_g, reg_W, reg_b
        a1W = (const float*)d_in[h0 + 0]; a1b = (const float*)d_in[h0 + 1];
        a2W = (const float*)d_in[h0 + 2]; a2b = (const float*)d_in[h0 + 3];
        ftW = (const float*)d_in[h0 + 4]; ftb = (const float*)d_in[h0 + 5];
        lnb = (const float*)d_in[h0 + 6]; lng = (const float*)d_in[h0 + 7];
        rW  = (const float*)d_in[h0 + 8]; rb  = (const float*)d_in[h0 + 9];
    }

    // --- workspace layout (256B-aligned carve-outs) ---
    size_t off = 0;
    auto alloc = [&](size_t bytes) -> void* {
        void* p = (char*)d_ws + off;
        off += (bytes + 255) & ~(size_t)255;
        return p;
    };
    float*          h    = (float*)alloc((size_t)NN * HID * 4);
    float*          xcur = (float*)alloc((size_t)NN * HID * 4);
    float*          agg  = (float*)alloc((size_t)NN * HID * 4);
    float*          as_  = (float*)alloc((size_t)NN * 4);
    float*          ad_  = (float*)alloc((size_t)NN * 4);
    unsigned*       menc = (unsigned*)alloc((size_t)NN * 4);
    float*          s    = (float*)alloc((size_t)NN * 4);
    float*          elog = (float*)alloc((size_t)NET * 4);
    float*          gsum = (float*)alloc((size_t)NG * HID * 4);
    float*          gcnt = (float*)alloc((size_t)NG * 4);
    unsigned short* xhi  = (unsigned short*)alloc((size_t)NN * HID * 2);
    unsigned short* xlo  = (unsigned short*)alloc((size_t)NN * HID * 2);
    unsigned short* whi  = (unsigned short*)alloc((size_t)HID * HID * 2);
    unsigned short* wlo  = (unsigned short*)alloc((size_t)HID * HID * 2);
    float*          out  = (float*)d_out;        // [pred(64) | feat(64*96)]

    const int TB = 256;
    for (int l = 0; l < 5; ++l) {
        const int K = (l == 0) ? FIN : HID;
        if (l == 0)
            k_split_x<<<cdiv(NN * FIN, TB), TB, 0, stream>>>(X, xhi, xlo);
        k_split_w<<<cdiv(HID * K, TB), TB, 0, stream>>>(cW[l], whi, wlo, K);
        if (l == 0)
            k_gemm<FIN><<<cdiv(NN / 16, 4), 128, 0, stream>>>(xhi, xlo, whi, wlo, h);
        else
            k_gemm<HID><<<cdiv(NN / 16, 4), 128, 0, stream>>>(xhi, xlo, whi, wlo, h);
        k_attn<<<cdiv(NN, 8), 256, 0, stream>>>(h, cAs[l], cAd[l], as_, ad_);
        k_init_layer<<<cdiv(NN * HID, TB), TB, 0, stream>>>(menc, s, agg);
        k_edge_max<<<cdiv(NET, TB), TB, 0, stream>>>(ei, as_, ad_, elog, menc);
        k_edge_exp<<<cdiv(NET, TB), TB, 0, stream>>>(ei, elog, menc, s);
        k_edge_norm<<<cdiv(NET, TB), TB, 0, stream>>>(ei, elog, s);
        k_edge_agg<<<cdiv(NET * HID, TB), TB, 0, stream>>>(ei, elog, h, agg);
        k_epilogue<<<cdiv(NN * HID, TB), TB, 0, stream>>>(agg, cB[l], bG[l], bB[l],
                                                          bRm[l], bRv[l], xcur, xhi, xlo);
    }
    k_zero_pool<<<cdiv(NG * HID, TB), TB, 0, stream>>>(gsum, gcnt);
    k_pool<<<cdiv(NN * HID, TB), TB, 0, stream>>>(xcur, batch, gsum, gcnt);
    k_head<<<NG, HID, 0, stream>>>(gsum, gcnt, ftW, ftb, lng, lnb,
                                   a1W, a1b, a2W, a2b, rW, rb, out);
}
